// SchNet_78623671320832
// MI455X (gfx1250) — compile-verified, matrix operand order checked by reading, not run on hardware
//
#include <hip/hip_runtime.h>
#include <hip/hip_bf16.h>

// ---------------------------------------------------------------------------
// SchNet on gfx1250: all 64x64 GEMMs via v_wmma_f32_16x16x32_f16.
// ef (E x 64 RBF features) is never materialized: it is synthesized directly
// into WMMA A-fragment registers from dist[E].  h / hv / agg (12.8 MB each)
// stay L2-resident; gather/scatter through f32 global atomics.
// ---------------------------------------------------------------------------

typedef __attribute__((ext_vector_type(16))) _Float16 v16h;
typedef __attribute__((ext_vector_type(8)))  _Float16 v8h;
typedef __attribute__((ext_vector_type(8)))  float    v8f;

#define DIM    64
#define NCONV  3
#define WPB    4            // 16-row tiles (waves) per block
#define TPB    (WPB * 32)   // wave32

static __device__ __forceinline__ float ssp(float x) {
  // shifted softplus: log(1+exp(x)) - log(2), numerically stable
  return fmaxf(x, 0.0f) + __logf(1.0f + __expf(-fabsf(x))) - 0.6931471805599453f;
}

static __device__ __forceinline__ v8f wmma32(v16h a, v16h b, v8f c) {
  // D = A(16x32 f16) * B(32x16 f16) + C(16x16 f32)
  return __builtin_amdgcn_wmma_f32_16x16x32_f16(false, a, false, b, (short)0, c,
                                                false, false);
}

// pack 8 contiguous f32 (16B-aligned) into halves a[lo..lo+7]
static __device__ __forceinline__ void pack8f(v16h& a, int lo, const float* p) {
  const float4* q = (const float4*)p;
  float4 u = q[0], v = q[1];
  a[lo + 0] = (_Float16)u.x; a[lo + 1] = (_Float16)u.y;
  a[lo + 2] = (_Float16)u.z; a[lo + 3] = (_Float16)u.w;
  a[lo + 4] = (_Float16)v.x; a[lo + 5] = (_Float16)v.y;
  a[lo + 6] = (_Float16)v.z; a[lo + 7] = (_Float16)v.w;
}

// pack 8 contiguous f16 (16B-aligned, LDS) into a[lo..lo+7]
static __device__ __forceinline__ void pack8h(v16h& a, int lo, const _Float16* p) {
  v8h u = *(const v8h*)p;
#pragma unroll
  for (int e = 0; e < 8; ++e) a[lo + e] = u[e];
}

// ---------------------------------------------------------------------------
__global__ void k_zero(float* p, long n) {
  long i = (long)blockIdx.x * blockDim.x + threadIdx.x;
  long s = (long)gridDim.x * blockDim.x;
  for (; i < n; i += s) p[i] = 0.0f;
}

__global__ void k_dist(const float* __restrict__ r, float* __restrict__ d, int E) {
  int i = blockIdx.x * blockDim.x + threadIdx.x;
  if (i >= E) return;
  float x = r[3 * i], y = r[3 * i + 1], z = r[3 * i + 2];
  d[i] = sqrtf(x * x + y * y + z * z);
}

__global__ void k_embed(const int* __restrict__ Z, const float* __restrict__ emb,
                        float* __restrict__ h, int V) {
  int i = blockIdx.x * blockDim.x + threadIdx.x;
  int v = i >> 4, c = i & 15;                 // 16 float4 per node row
  if (v >= V) return;
  ((float4*)h)[(size_t)v * 16 + c] = ((const float4*)emb)[(size_t)Z[v] * 16 + c];
}

// Pre-pack a 64x64 f32 weight (row = K, col = N) into the per-lane B-fragment
// order of v_wmma_f32_16x16x32_f16 (ISA 7.12.2, 16-bit 16x32 layout), one
// contiguous 16-half record per (kstep s, ntile t, lane).
__global__ void k_pack(const float* Wn, const float* We1, const float* We2,
                       const float* Wc, const float* Wp, _Float16* wfrag) {
  int y = blockIdx.y;                 // 0..14 : conv*5 + which
  int conv = y / 5, which = y % 5;
  const float* W;
  switch (which) {
    case 0:  W = Wn;  break;
    case 1:  W = We1; break;
    case 2:  W = We2; break;
    case 3:  W = Wc;  break;
    default: W = Wp;  break;
  }
  W += (size_t)conv * DIM * DIM;
  int i = blockIdx.x * blockDim.x + threadIdx.x;
  if (i >= 4096) return;
  int e = i & 15, lane = (i >> 4) & 31, frag = i >> 9;    // frag = s*4 + t
  int s = frag >> 2, t = frag & 3, half = lane >> 4;
  int kk = 32 * s + ((e < 8) ? (8 * half + e) : (16 + 8 * half + (e - 8)));
  int n  = 16 * t + (lane & 15);
  wfrag[(size_t)y * 4096 + i] = (_Float16)W[kk * DIM + n];
}

// ---------------------------------------------------------------------------
// hv = h @ Wn + bn          (one 16-row tile per wave)
__global__ __launch_bounds__(TPB)
void k_node_proj(const float* __restrict__ h, float* __restrict__ hv,
                 const _Float16* __restrict__ wf, const float* __restrict__ bias,
                 int V) {
  __shared__ __align__(32) _Float16 lw[4096];
  {
    const uint4* s = (const uint4*)wf;
    uint4* d = (uint4*)lw;
    for (int i = threadIdx.x; i < 512; i += blockDim.x) d[i] = s[i];
  }
  __syncthreads();
  int wave = threadIdx.x >> 5, lane = threadIdx.x & 31;
  long base = ((long)blockIdx.x * WPB + wave) * 16;
  if (base >= V) return;                     // wave-uniform: EXEC stays all-1s
  int m = lane & 15, half = lane >> 4;

  const float* row = h + (size_t)(base + m) * DIM;
  v16h a0, a1;
  pack8f(a0, 0, row + 8 * half);       pack8f(a0, 8, row + 16 + 8 * half);
  pack8f(a1, 0, row + 32 + 8 * half);  pack8f(a1, 8, row + 48 + 8 * half);

  const v16h* W = (const v16h*)lw;
#pragma unroll
  for (int t = 0; t < 4; ++t) {
    v8f c = {};
    c = wmma32(a0, W[t * 32 + lane], c);
    c = wmma32(a1, W[(4 + t) * 32 + lane], c);
    int n = 16 * t + m;
    float b = bias[n];
#pragma unroll
    for (int r2 = 0; r2 < 8; ++r2)
      hv[(size_t)(base + r2 + 8 * half) * DIM + n] = c[r2] + b;
  }
}

// ---------------------------------------------------------------------------
// Fused edge pipeline: RBF -> GEMM -> ssp -> GEMM -> ssp -> gather*mul -> scatter
__global__ __launch_bounds__(TPB)
void k_edge(const float* __restrict__ dist, const int* __restrict__ src,
            const int* __restrict__ dst, const float* __restrict__ hv,
            float* __restrict__ agg,
            const _Float16* __restrict__ w1f, const _Float16* __restrict__ w2f,
            const float* __restrict__ be1, const float* __restrict__ be2, int E) {
  __shared__ __align__(32) _Float16 lw1[4096];
  __shared__ __align__(32) _Float16 lw2[4096];
  __shared__ __align__(32) _Float16 hbuf[WPB][16 * DIM];
  {
    const uint4* s1 = (const uint4*)w1f;
    const uint4* s2 = (const uint4*)w2f;
    uint4* d1 = (uint4*)lw1;
    uint4* d2 = (uint4*)lw2;
    for (int i = threadIdx.x; i < 512; i += blockDim.x) { d1[i] = s1[i]; d2[i] = s2[i]; }
  }
  __syncthreads();
  int wave = threadIdx.x >> 5, lane = threadIdx.x & 31;
  long base = ((long)blockIdx.x * WPB + wave) * 16;
  if (base >= E) return;
  int m = lane & 15, half = lane >> 4;

  // RBF expansion straight into A-fragment registers (ef never hits memory)
  const float step  = 5.0f / 63.0f;            // CUTOFF/(NUM_G-1)
  const float coeff = -0.5f / (step * step);
  float d0 = dist[base + m];
  v16h a0, a1;
#pragma unroll
  for (int e = 0; e < 16; ++e) {
    int k0 = (e < 8) ? (8 * half + e) : (16 + 8 * half + (e - 8));
    float t0 = d0 - step * (float)k0;
    float t1 = d0 - step * (float)(k0 + 32);
    a0[e] = (_Float16)__expf(coeff * t0 * t0);
    a1[e] = (_Float16)__expf(coeff * t1 * t1);
  }

  const v16h* W1 = (const v16h*)lw1;
  const v16h* W2 = (const v16h*)lw2;
  _Float16* hb = hbuf[wave];

  // he1 = ssp(ef @ We1 + be1) -> f16 tile in LDS (C-layout -> row-major)
#pragma unroll
  for (int t = 0; t < 4; ++t) {
    v8f c = {};
    c = wmma32(a0, W1[t * 32 + lane], c);
    c = wmma32(a1, W1[(4 + t) * 32 + lane], c);
    int n = 16 * t + m;
    float b = be1[n];
#pragma unroll
    for (int r2 = 0; r2 < 8; ++r2)
      hb[(r2 + 8 * half) * DIM + n] = (_Float16)ssp(c[r2] + b);
  }

  // re-load as A fragments (same-wave LDS RAW: compiler inserts s_wait_dscnt)
  v16h b0, b1;
  const _Float16* hr = hb + m * DIM;
  pack8h(b0, 0, hr + 8 * half);       pack8h(b0, 8, hr + 16 + 8 * half);
  pack8h(b1, 0, hr + 32 + 8 * half);  pack8h(b1, 8, hr + 48 + 8 * half);

  int sIdx[8], dIdx[8];
#pragma unroll
  for (int r2 = 0; r2 < 8; ++r2) {
    sIdx[r2] = src[base + r2 + 8 * half];
    dIdx[r2] = dst[base + r2 + 8 * half];
  }

  // he2 = ssp(he1 @ We2 + be2); m = hv[src]*he2; agg[dst] += m  (L2-resident)
#pragma unroll
  for (int t = 0; t < 4; ++t) {
    v8f c = {};
    c = wmma32(b0, W2[t * 32 + lane], c);
    c = wmma32(b1, W2[(4 + t) * 32 + lane], c);
    int n = 16 * t + m;
    float b = be2[n];
#pragma unroll
    for (int r2 = 0; r2 < 8; ++r2) {
      float he = ssp(c[r2] + b);
      float mv = hv[(size_t)sIdx[r2] * DIM + n] * he;
      atomicAdd(&agg[(size_t)dIdx[r2] * DIM + n], mv);
    }
  }
}

// ---------------------------------------------------------------------------
// h = ssp(agg @ Wc + bc) @ Wp + bp ; also clears agg for the next conv
__global__ __launch_bounds__(TPB)
void k_node_out(float* __restrict__ agg, float* __restrict__ h,
                const _Float16* __restrict__ wcf, const _Float16* __restrict__ wpf,
                const float* __restrict__ bc, const float* __restrict__ bp, int V) {
  __shared__ __align__(32) _Float16 lwc[4096];
  __shared__ __align__(32) _Float16 lwp[4096];
  __shared__ __align__(32) _Float16 hbuf[WPB][16 * DIM];
  {
    const uint4* s1 = (const uint4*)wcf;
    const uint4* s2 = (const uint4*)wpf;
    uint4* d1 = (uint4*)lwc;
    uint4* d2 = (uint4*)lwp;
    for (int i = threadIdx.x; i < 512; i += blockDim.x) { d1[i] = s1[i]; d2[i] = s2[i]; }
  }
  __syncthreads();
  int wave = threadIdx.x >> 5, lane = threadIdx.x & 31;
  long base = ((long)blockIdx.x * WPB + wave) * 16;
  if (base >= V) return;
  int m = lane & 15, half = lane >> 4;

  float* row = agg + (size_t)(base + m) * DIM;
  v16h a0, a1;
  pack8f(a0, 0, row + 8 * half);       pack8f(a0, 8, row + 16 + 8 * half);
  pack8f(a1, 0, row + 32 + 8 * half);  pack8f(a1, 8, row + 48 + 8 * half);
  {
    // consume-and-clear (each element is loaded by exactly one lane)
    float4 z = make_float4(0.f, 0.f, 0.f, 0.f);
    float4* q = (float4*)row;
    int o0 = (8 * half) >> 2;
    q[o0] = z;      q[o0 + 1] = z;
    q[o0 + 4] = z;  q[o0 + 5] = z;
    q[o0 + 8] = z;  q[o0 + 9] = z;
    q[o0 + 12] = z; q[o0 + 13] = z;
  }

  const v16h* WC = (const v16h*)lwc;
  const v16h* WP = (const v16h*)lwp;
  _Float16* hb = hbuf[wave];
#pragma unroll
  for (int t = 0; t < 4; ++t) {
    v8f c = {};
    c = wmma32(a0, WC[t * 32 + lane], c);
    c = wmma32(a1, WC[(4 + t) * 32 + lane], c);
    int n = 16 * t + m;
    float b = bc[n];
#pragma unroll
    for (int r2 = 0; r2 < 8; ++r2)
      hb[(r2 + 8 * half) * DIM + n] = (_Float16)ssp(c[r2] + b);
  }

  v16h b0, b1;
  const _Float16* hr = hb + m * DIM;
  pack8h(b0, 0, hr + 8 * half);       pack8h(b0, 8, hr + 16 + 8 * half);
  pack8h(b1, 0, hr + 32 + 8 * half);  pack8h(b1, 8, hr + 48 + 8 * half);

#pragma unroll
  for (int t = 0; t < 4; ++t) {
    v8f c = {};
    c = wmma32(b0, WP[t * 32 + lane], c);
    c = wmma32(b1, WP[(4 + t) * 32 + lane], c);
    int n = 16 * t + m;
    float b = bp[n];
#pragma unroll
    for (int r2 = 0; r2 < 8; ++r2)
      h[(size_t)(base + r2 + 8 * half) * DIM + n] = c[r2] + b;
  }
}

// ---------------------------------------------------------------------------
// head MLP (64->32->1) + per-graph sum/count
__global__ __launch_bounds__(128)
void k_head(const float* __restrict__ h, const int* __restrict__ n2g,
            const float* __restrict__ Wd1, const float* __restrict__ bd1,
            const float* __restrict__ Wd2, const float* __restrict__ bd2,
            float* __restrict__ gsum, float* __restrict__ gcnt, int V) {
  __shared__ float w1[DIM * 32];
  __shared__ float w2[32];
  __shared__ float b1[32];
  for (int i = threadIdx.x; i < DIM * 32; i += blockDim.x) w1[i] = Wd1[i];
  if (threadIdx.x < 32) { w2[threadIdx.x] = Wd2[threadIdx.x]; b1[threadIdx.x] = bd1[threadIdx.x]; }
  __syncthreads();
  int v = blockIdx.x * blockDim.x + threadIdx.x;
  if (v >= V) return;
  float hr[DIM];
#pragma unroll
  for (int k = 0; k < DIM; ++k) hr[k] = h[(size_t)v * DIM + k];
  float res = bd2[0];
#pragma unroll 4
  for (int j = 0; j < 32; ++j) {
    float acc = b1[j];
#pragma unroll
    for (int k = 0; k < DIM; ++k) acc = fmaf(hr[k], w1[k * 32 + j], acc);
    res = fmaf(ssp(acc), w2[j], res);
  }
  int g = n2g[v];
  atomicAdd(&gsum[g], res);
  atomicAdd(&gcnt[g], 1.0f);
}

__global__ void k_finalize(const float* gsum, const float* gcnt, float* out, int G) {
  int g = blockIdx.x * blockDim.x + threadIdx.x;
  if (g >= G) return;
  out[g] = gsum[g] / fmaxf(gcnt[g], 1.0f);
}

// ---------------------------------------------------------------------------
extern "C" void kernel_launch(void* const* d_in, const int* in_sizes, int n_in,
                              void* d_out, int out_size, void* d_ws, size_t ws_size,
                              hipStream_t stream) {
  const int*   Z   = (const int*)d_in[0];
  const int*   src = (const int*)d_in[1];
  const int*   dst = (const int*)d_in[2];
  const int*   n2g = (const int*)d_in[3];
  const float* r   = (const float*)d_in[4];
  // d_in[5] = n_graphs scalar on device; use out_size instead
  const float* emb = (const float*)d_in[6];
  const float* Wn  = (const float*)d_in[7];
  const float* bn  = (const float*)d_in[8];
  const float* We1 = (const float*)d_in[9];
  const float* be1 = (const float*)d_in[10];
  const float* We2 = (const float*)d_in[11];
  const float* be2 = (const float*)d_in[12];
  const float* Wc  = (const float*)d_in[13];
  const float* bc  = (const float*)d_in[14];
  const float* Wp  = (const float*)d_in[15];
  const float* bp  = (const float*)d_in[16];
  const float* Wd1 = (const float*)d_in[17];
  const float* bd1 = (const float*)d_in[18];
  const float* Wd2 = (const float*)d_in[19];
  const float* bd2 = (const float*)d_in[20];
  const int V = in_sizes[0];
  const int E = in_sizes[1];
  const int G = out_size;
  (void)n_in; (void)ws_size;

  char* ws = (char*)d_ws;
  size_t off = 0;
  auto alloc = [&](size_t bytes) -> char* {
    char* p = ws + off;
    off += (bytes + 255) & ~(size_t)255;
    return p;
  };
  float*    distb = (float*)alloc((size_t)E * 4);
  float*    hnode = (float*)alloc((size_t)V * DIM * 4);
  float*    hvbuf = (float*)alloc((size_t)V * DIM * 4);
  float*    aggb  = (float*)alloc((size_t)V * DIM * 4);
  _Float16* wfrag = (_Float16*)alloc((size_t)NCONV * 5 * 4096 * 2);
  float*    gsum  = (float*)alloc((size_t)G * 4);
  float*    gcnt  = (float*)alloc((size_t)G * 4);

  // deterministic init of accumulators every call
  k_zero<<<2048, 256, 0, stream>>>(aggb, (long)V * DIM);
  k_zero<<<8, 256, 0, stream>>>(gsum, G);
  k_zero<<<8, 256, 0, stream>>>(gcnt, G);

  k_dist<<<(E + 255) / 256, 256, 0, stream>>>(r, distb, E);
  k_embed<<<((V * 16) + 255) / 256, 256, 0, stream>>>(Z, emb, hnode, V);
  dim3 pg(16, NCONV * 5);
  k_pack<<<pg, 256, 0, stream>>>(Wn, We1, We2, Wc, Wp, wfrag);

  int nBlocksV = (((V + 15) / 16) + WPB - 1) / WPB;
  int nBlocksE = (((E + 15) / 16) + WPB - 1) / WPB;
  for (int i = 0; i < NCONV; ++i) {
    const _Float16* f = wfrag + (size_t)i * 5 * 4096;
    k_node_proj<<<nBlocksV, TPB, 0, stream>>>(hnode, hvbuf, f + 0 * 4096,
                                              bn + i * DIM, V);
    k_edge<<<nBlocksE, TPB, 0, stream>>>(distb, src, dst, hvbuf, aggb,
                                         f + 1 * 4096, f + 2 * 4096,
                                         be1 + i * DIM, be2 + i * DIM, E);
    k_node_out<<<nBlocksV, TPB, 0, stream>>>(aggb, hnode, f + 3 * 4096,
                                             f + 4 * 4096, bc + i * DIM,
                                             bp + i * DIM, V);
  }
  k_head<<<(V + 127) / 128, 128, 0, stream>>>(hnode, n2g, Wd1, bd1, Wd2, bd2,
                                              gsum, gcnt, V);
  k_finalize<<<(G + 255) / 256, 256, 0, stream>>>(gsum, gcnt, (float*)d_out, G);
}
